// unit_gcn_6322191860404
// MI455X (gfx1250) — matrix operand order repaired and checked by reading, not compiled, not run on hardware
//
#include <hip/hip_runtime.h>
#include <hip/hip_bf16.h>

#define NN 64
#define CC 64
#define TT 256
#define VV 25
#define SS 3
#define OO 64
#define RR 8
#define BN_EPS 1e-5f

typedef __attribute__((ext_vector_type(16))) _Float16 v16h;
typedef __attribute__((ext_vector_type(8)))  float    v8f;

union V16H { v16h v; _Float16 h[16]; uint4 u4[2]; };
union V8F  { v8f  v; float f[8]; };

// ---------------------------------------------------------------- k0: zero BN accumulators
__global__ void k_zero(float* bn) {
    if (threadIdx.x < 128) bn[threadIdx.x] = 0.f;
}

// ---------------------------------------------------------------- k1: xm = mean_t x  (N,C,V)
__global__ __launch_bounds__(256) void k_mean(const float* __restrict__ x,
                                              float* __restrict__ xm) {
    int g = blockIdx.x * 256 + threadIdx.x;
    if (g >= NN * CC * VV) return;
    int n = g / (CC * VV);
    int rem = g - n * (CC * VV);
    int c = rem / VV, v = rem - c * VV;
    const float* p = x + ((size_t)(n * CC + c)) * TT * VV + v;
    float s = 0.f;
#pragma unroll 8
    for (int t = 0; t < TT; ++t) s += p[t * VV];
    xm[g] = s * (1.0f / TT);
}

// ---------------------------------------------------------------- k2: adjacency (f16, padded 32x32, [s][n][o][u][v])
__global__ __launch_bounds__(256) void k_adj(const float* __restrict__ xm,
                                             const float* __restrict__ Ag,
                                             const float* __restrict__ alphap,
                                             const float* __restrict__ w1,
                                             const float* __restrict__ b1,
                                             const float* __restrict__ w2,
                                             const float* __restrict__ b2,
                                             const float* __restrict__ w4,
                                             const float* __restrict__ b4,
                                             _Float16* __restrict__ adjh) {
    __shared__ float x1L[RR][VV], x2L[RR][VV];
    int s = blockIdx.x / NN, n = blockIdx.x - s * NN;
    int tid = threadIdx.x;
    if (tid < RR * VV) {
        int r = tid / VV, v = tid - r * VV;
        const float* pxm = xm + n * CC * VV + v;
        const float* pw1 = w1 + (s * RR + r) * CC;
        const float* pw2 = w2 + (s * RR + r) * CC;
        float a1 = b1[s * RR + r], a2 = b2[s * RR + r];
#pragma unroll 8
        for (int c = 0; c < CC; ++c) {
            float xv = pxm[c * VV];
            a1 += pw1[c] * xv;
            a2 += pw2[c] * xv;
        }
        x1L[r][v] = a1;
        x2L[r][v] = a2;
    }
    __syncthreads();
    float alpha = alphap[0];
    size_t base = ((size_t)(s * NN + n)) * (OO * 32 * 32);
    for (int e = tid; e < OO * 32 * 32; e += 256) {
        int o = e >> 10;
        int uu = (e >> 5) & 31;
        int vv2 = e & 31;
        float val = 0.f;
        if (uu < VV && vv2 < VV) {
            float acc = 0.f;
#pragma unroll
            for (int r = 0; r < RR; ++r)
                acc += w4[(s * OO + o) * RR + r] * tanhf(x1L[r][uu] - x2L[r][vv2]);
            val = (acc + b4[s * OO + o]) * alpha + Ag[s * VV * VV + uu * VV + vv2];
        }
        adjh[base + e] = (_Float16)val;
    }
}

// ---------------------------------------------------------------- k2b: xt = f16 transpose of x -> [n][t][v][c]
__global__ __launch_bounds__(256) void k_xt(const float* __restrict__ x,
                                            _Float16* __restrict__ xt) {
    __shared__ float tile[CC * VV]; // [c][v]
    int n = blockIdx.x / TT, t = blockIdx.x - n * TT;
    int tid = threadIdx.x;
    for (int e = tid; e < CC * VV; e += 256) {
        int c = e / VV, v = e - c * VV;
        tile[e] = x[(((size_t)(n * CC + c)) * TT + t) * VV + v];
    }
    __syncthreads();
    size_t ob = ((size_t)(n * TT + t)) * VV * CC;
    for (int e = tid; e < VV * CC; e += 256) {
        int v = e >> 6, c = e & 63;
        xt[ob + e] = (_Float16)tile[c * VV + v];
    }
}

// ---------------------------------------------------------------- k3: fused WMMA main kernel
// grid = N * (T/16); 256 threads = 8 waves. Per block: 16 time steps.
__global__ __launch_bounds__(256) void k_main(const float* __restrict__ w3g,
                                              const float* __restrict__ b3g,
                                              const _Float16* __restrict__ xt,
                                              const _Float16* __restrict__ adjh,
                                              float* __restrict__ y_ws,
                                              float* __restrict__ bn) {
    __shared__ __align__(16) _Float16 x3h[32 * 16 * 32]; // [o_loc<32][t<16][v<32] : 32 KB
    const int blk = blockIdx.x;
    const int n = blk >> 4;
    const int tblk = blk & 15;
    const int tid = threadIdx.x;
    const int lane = tid & 31, wave = tid >> 5;
    const int l16 = lane & 15, half = lane >> 4;

    // zero once: v-pad (25..31) must stay 0 (K=32 padding), stage1 only rewrites v<25
    for (int i = tid; i < 32 * 16 * 32 / 2; i += 256) ((unsigned int*)x3h)[i] = 0u;
    __syncthreads();

    const size_t xtb = ((size_t)(n * TT) + (size_t)tblk * 16) * (VV * CC);

    for (int oh = 0; oh < 2; ++oh) {
        V8F acc[8];
#pragma unroll
        for (int j = 0; j < 8; ++j)
#pragma unroll
            for (int r = 0; r < 8; ++r) acc[j].f[r] = 0.f;

        for (int s = 0; s < SS; ++s) {
            // -------- stage 1: x3 = w3 @ x (+b3), for o-half oh, into LDS (f16) --------
            const int otile = oh * 2 + (wave & 1); // absolute 16-o tile
            V16H afr[2];
#pragma unroll
            for (int ks = 0; ks < 2; ++ks) {
                const int kb = ks * 32;
                const int o = otile * 16 + l16; // A: M = lane%16
                const float* pw = w3g + (size_t)(s * OO + o) * CC;
#pragma unroll
                for (int ch = 0; ch < 2; ++ch) {
                    const int c0 = kb + ch * 16 + half * 8; // A K chunks: {h*8..}, {16+h*8..}
                    const float4 f0 = *reinterpret_cast<const float4*>(pw + c0);
                    const float4 f1 = *reinterpret_cast<const float4*>(pw + c0 + 4);
                    afr[ks].h[ch * 8 + 0] = (_Float16)f0.x;
                    afr[ks].h[ch * 8 + 1] = (_Float16)f0.y;
                    afr[ks].h[ch * 8 + 2] = (_Float16)f0.z;
                    afr[ks].h[ch * 8 + 3] = (_Float16)f0.w;
                    afr[ks].h[ch * 8 + 4] = (_Float16)f1.x;
                    afr[ks].h[ch * 8 + 5] = (_Float16)f1.y;
                    afr[ks].h[ch * 8 + 6] = (_Float16)f1.z;
                    afr[ks].h[ch * 8 + 7] = (_Float16)f1.w;
                }
            }
            float b3r[8];
#pragma unroll
            for (int r = 0; r < 8; ++r) b3r[r] = b3g[s * OO + otile * 16 + r + half * 8];

            for (int tvt = (wave >> 1); tvt < 25; tvt += 4) { // 25 tv-tiles of 16 (=400)
                const int tv = tvt * 16 + l16;                // B: N = lane%16
                const _Float16* px = xt + xtb + (size_t)tv * CC;
                __builtin_prefetch(px + 4 * 16 * CC, 0, 0);   // global_prefetch_b8
                V16H bfr[2];
#pragma unroll
                for (int ks = 0; ks < 2; ++ks) {
                    const int c0 = ks * 32 + half * 8;        // B K chunks: {h*8..}, {16+h*8..}
                    bfr[ks].u4[0] = *reinterpret_cast<const uint4*>(px + c0);
                    bfr[ks].u4[1] = *reinterpret_cast<const uint4*>(px + c0 + 16);
                }
                V8F d;
#pragma unroll
                for (int r = 0; r < 8; ++r) d.f[r] = b3r[r];
                d.v = __builtin_amdgcn_wmma_f32_16x16x32_f16(false, afr[0].v, false, bfr[0].v,
                                                             (short)0, d.v, false, false);
                d.v = __builtin_amdgcn_wmma_f32_16x16x32_f16(false, afr[1].v, false, bfr[1].v,
                                                             (short)0, d.v, false, false);
                const int tt2 = tv / VV, vv2 = tv - tt2 * VV;
                const int obase = (otile - oh * 2) * 16;
#pragma unroll
                for (int r = 0; r < 8; ++r) {                 // D: row = r + 8*half
                    const int o_loc = obase + r + half * 8;
                    x3h[(o_loc * 16 + tt2) * 32 + vv2] = (_Float16)d.f[r];
                }
            }
            __syncthreads();

            // -------- stage 2: z[t,u] += x3[t,v] @ adj[u,v]^T  (K=32 covers v) --------
#pragma unroll
            for (int j = 0; j < 8; ++j) {
                const int g = wave + 8 * j;       // 64 tiles: 32 o_loc x 2 u-tiles
                const int o_loc = g >> 1, ut = g & 1;
                V16H af;                          // A: rows = t
                const _Float16* pa = &x3h[(o_loc * 16 + l16) * 32];
                af.u4[0] = *reinterpret_cast<const uint4*>(pa + half * 8);
                af.u4[1] = *reinterpret_cast<const uint4*>(pa + 16 + half * 8);
                const int o = oh * 32 + o_loc;
                const int u = ut * 16 + l16;      // B: cols = u
                const _Float16* pb = adjh + ((((size_t)s * NN + n) * OO + o) * 32 + u) * 32;
                V16H bf;
                bf.u4[0] = *reinterpret_cast<const uint4*>(pb + half * 8);
                bf.u4[1] = *reinterpret_cast<const uint4*>(pb + 16 + half * 8);
                acc[j].v = __builtin_amdgcn_wmma_f32_16x16x32_f16(false, af.v, false, bf.v,
                                                                  (short)0, acc[j].v, false, false);
            }
            __syncthreads(); // before next s's stage1 overwrites x3h
        } // s

        // -------- writeout (pre-BN y) + BN partial moments --------
#pragma unroll
        for (int j = 0; j < 8; ++j) {
            const int g = wave + 8 * j;
            const int o_loc = g >> 1, ut = g & 1;
            const int o = oh * 32 + o_loc;
            const int u = ut * 16 + l16;
            float s1 = 0.f, s2 = 0.f;
            if (u < VV) {
#pragma unroll
                for (int r = 0; r < 8; ++r) {
                    const int t = r + half * 8;
                    const float vv3 = acc[j].f[r];
                    y_ws[((size_t)(n * OO + o) * TT + tblk * 16 + t) * VV + u] = vv3;
                    s1 += vv3;
                    s2 += vv3 * vv3;
                }
            }
#pragma unroll
            for (int off = 16; off > 0; off >>= 1) {
                s1 += __shfl_xor(s1, off, 32);
                s2 += __shfl_xor(s2, off, 32);
            }
            if (lane == 0) {
                atomicAdd(&bn[o], s1);
                atomicAdd(&bn[64 + o], s2);
            }
        }
    } // oh
}

// ---------------------------------------------------------------- k4: BN finalize -> scale/shift
__global__ void k_bnfin(float* bn, const float* __restrict__ gamma,
                        const float* __restrict__ beta) {
    int o = threadIdx.x;
    if (o < OO) {
        const float cnt = (float)(NN * TT * VV);
        float mu = bn[o] / cnt;
        float var = bn[64 + o] / cnt - mu * mu;
        float sc = gamma[o] * rsqrtf(var + BN_EPS);
        bn[128 + o] = sc;
        bn[192 + o] = beta[o] - mu * sc;
    }
}

// ---------------------------------------------------------------- k5: out = relu(scale*y + shift + x)
__global__ __launch_bounds__(256) void k_epilogue(const float* __restrict__ y_ws,
                                                  const float* __restrict__ x,
                                                  const float* __restrict__ bn,
                                                  float* __restrict__ out) {
    int i4 = blockIdx.x * 256 + threadIdx.x;
    if (i4 >= NN * OO * TT * VV / 4) return;
    int o = (i4 / (TT * VV / 4)) & (OO - 1);
    float sc = bn[128 + o], sh = bn[192 + o];
    float4 yv = reinterpret_cast<const float4*>(y_ws)[i4];
    float4 xv = reinterpret_cast<const float4*>(x)[i4];
    float4 r;
    r.x = fmaxf(fmaf(yv.x, sc, sh) + xv.x, 0.f);
    r.y = fmaxf(fmaf(yv.y, sc, sh) + xv.y, 0.f);
    r.z = fmaxf(fmaf(yv.z, sc, sh) + xv.z, 0.f);
    r.w = fmaxf(fmaf(yv.w, sc, sh) + xv.w, 0.f);
    reinterpret_cast<float4*>(out)[i4] = r;
}

// ---------------------------------------------------------------- launcher
extern "C" void kernel_launch(void* const* d_in, const int* in_sizes, int n_in,
                              void* d_out, int out_size, void* d_ws, size_t ws_size,
                              hipStream_t stream) {
    const float* x     = (const float*)d_in[0];
    const float* A     = (const float*)d_in[1];
    const float* alpha = (const float*)d_in[2];
    const float* w1    = (const float*)d_in[3];
    const float* b1    = (const float*)d_in[4];
    const float* w2    = (const float*)d_in[5];
    const float* b2    = (const float*)d_in[6];
    const float* w3    = (const float*)d_in[7];
    const float* b3    = (const float*)d_in[8];
    const float* w4    = (const float*)d_in[9];
    const float* b4    = (const float*)d_in[10];
    const float* gamma = (const float*)d_in[11];
    const float* beta  = (const float*)d_in[12];
    float* out = (float*)d_out;

    char* ws = (char*)d_ws;
    // workspace layout (16B-aligned blocks)
    float*     bn   = (float*)ws;                               // 256 f32: sum,sumsq,scale,shift
    float*     xm   = (float*)(ws + 1024);                      // 102400 f32
    _Float16*  adjh = (_Float16*)(ws + 1024 + 409600);          // 3*64*64*1024 f16 = 25.2 MB
    _Float16*  xtb  = (_Float16*)(ws + 25576448);               // 64*256*25*64 f16 = 52.4 MB
    float*     y_ws = (float*)(ws + 78005248);                  // 26.2M f32 = 104.9 MB

    k_zero<<<1, 128, 0, stream>>>(bn);
    k_mean<<<(NN * CC * VV + 255) / 256, 256, 0, stream>>>(x, xm);
    k_adj<<<SS * NN, 256, 0, stream>>>(xm, A, alpha, w1, b1, w2, b2, w4, b4, adjh);
    k_xt<<<NN * TT, 256, 0, stream>>>(x, xtb);
    k_main<<<NN * (TT / 16), 256, 0, stream>>>(w3, b3, xtb, adjh, y_ws, bn);
    k_bnfin<<<1, 64, 0, stream>>>(bn, gamma, beta);
    k_epilogue<<<(NN * OO * TT * VV / 4 + 255) / 256, 256, 0, stream>>>(y_ws, x, bn, out);
}
